// DyChannel_25177098289279
// MI455X (gfx1250) — matrix operand order repaired and checked by reading, not compiled, not run on hardware
//
#include <hip/hip_runtime.h>
#include <hip/hip_bf16.h>

// ---------------------------------------------------------------------------
// DyChannel MoE conv for gfx1250 (MI455X).
//   out[b] = x[b] (NCHW f32, 16x256x64x64)  conv3x3,pad1  w[b],
//   w[b,o,c,dy,dx] = sum_k sigmoid(route(x))[b,k,c] * EW[k,o,c,dy,dx]
// Experts are collapsed in weight space (77 GFLOP main conv), executed as an
// implicit GEMM on v_wmma_f32_16x16x32_bf16. x is staged as zero-padded HWC
// bf16 so the WMMA K dimension (input channel) is contiguous; combined
// weights are staged pre-swizzled in A-fragment order so A loads are raw
// global_load_b128. Each wave owns a 2Mx2N register tile (4 accumulators),
// giving 2 b128 loads per WMMA.
// ---------------------------------------------------------------------------

typedef __attribute__((ext_vector_type(16))) __bf16 v16bf;
typedef __attribute__((ext_vector_type(8)))  __bf16 v8bf;
typedef __attribute__((ext_vector_type(8)))  float  v8f;

union BF16Frag { v16bf v; v8bf h[2]; };

__device__ __forceinline__ unsigned short f2bf(float f) {
  unsigned int u = __float_as_uint(f);
  u = (u + 0x7FFFu + ((u >> 16) & 1u)) >> 16;   // round-to-nearest-even
  return (unsigned short)u;
}

// ------------------------------ Phase 1a: 4x4 adaptive avg pool -------------
__global__ __launch_bounds__(256)
void pool_kernel(const float* __restrict__ x, float* __restrict__ pooled) {
  const int bc = blockIdx.x;                       // b*256 + c
  __shared__ float bins[16];
  if (threadIdx.x < 16) bins[threadIdx.x] = 0.f;
  __syncthreads();
  const float* xb = x + (size_t)bc * 4096;
  float part[4] = {0.f, 0.f, 0.f, 0.f};
  #pragma unroll
  for (int i = 0; i < 16; ++i)                     // pixel p = tid + i*256
    part[i >> 2] += xb[threadIdx.x + (i << 8)];    // row-bin = i>>2
  const int cj = (threadIdx.x & 63) >> 4;          // col-bin
  #pragma unroll
  for (int ri = 0; ri < 4; ++ri) atomicAdd(&bins[ri * 4 + cj], part[ri]);
  __syncthreads();
  if (threadIdx.x < 16)
    pooled[(size_t)bc * 16 + threadIdx.x] = bins[threadIdx.x] * (1.f / 256.f);
}

// ------------------------------ Phase 1b: routing MLP -----------------------
__global__ __launch_bounds__(256)
void route_kernel(const float* __restrict__ pooled,
                  const float* __restrict__ rconv_w, const float* __restrict__ rconv_b,
                  const float* __restrict__ fc_w,    const float* __restrict__ fc_b,
                  float* __restrict__ rbuf) {
  const int b = blockIdx.x;
  __shared__ float y1[256];
  const int tid = threadIdx.x;
  const int ri = tid >> 4, ij = tid & 15;          // (reduction ch, spatial)
  const float* pb = pooled + (size_t)b * 4096;     // [c][16]
  float s = rconv_b[ri];
  for (int c = 0; c < 256; ++c)
    s += pb[c * 16 + ij] * rconv_w[ri * 256 + c];
  y1[tid] = fmaxf(s, 0.f);                         // flatten order (rc, i, j)
  __syncthreads();
  for (int e = tid; e < 768; e += 256) {
    float t = fc_b[e];
    const float* fw = fc_w + (size_t)e * 256;
    for (int i = 0; i < 256; ++i) t += y1[i] * fw[i];
    t = fmaxf(t, 0.f);
    rbuf[(size_t)b * 768 + e] = 1.f / (1.f + expf(-t));   // [b][k*256+c]
  }
}

// ------------------------------ Phase 2a: zero padded x ---------------------
__global__ __launch_bounds__(256)
void zero_kernel(uint4* __restrict__ p, size_t n16) {
  const uint4 z = {0u, 0u, 0u, 0u};
  for (size_t i = (size_t)blockIdx.x * 256 + threadIdx.x; i < n16;
       i += (size_t)gridDim.x * 256)
    p[i] = z;
}

// ------------------------------ Phase 2b: x CHW -> padded HWC bf16 ----------
// xp[b][yp(66)][wp(66)][c(256)], interior = bf16(x), border stays zero.
__global__ __launch_bounds__(256)
void packx_kernel(const float* __restrict__ x, unsigned short* __restrict__ xp) {
  const int blk = blockIdx.x;
  const int y = blk & 63, b = blk >> 6;
  __shared__ unsigned short tile[64][65];          // [w][c-chunk], pad bank
  const float* xb = x + ((size_t)b * 256 * 64 + y) * 64;    // + c*4096 + w
  unsigned short* xpb = xp + ((size_t)(b * 66 + (y + 1))) * 66 * 256;
  const int tid = threadIdx.x;
  for (int c0 = 0; c0 < 256; c0 += 64) {
    const int w = tid & 63, cq = tid >> 6;
    #pragma unroll
    for (int ci = cq; ci < 64; ci += 4)            // coalesced read along w
      tile[w][ci] = f2bf(xb[(size_t)(c0 + ci) * 4096 + w]);
    __syncthreads();
    const int c = tid & 63, wq = tid >> 6;
    #pragma unroll
    for (int wi = wq; wi < 64; wi += 4)            // coalesced write along c
      xpb[(size_t)(wi + 1) * 256 + c0 + c] = tile[wi][c];
    __syncthreads();
  }
}

// ------------------------------ Phase 2c: combine + A-fragment pack ---------
// wpack[b][t(9)][cc(8)][MT(16)][lane(32)][slot(16)] bf16, where for the
// 16-bit 16x32 A fragment: m = lane&15, kk = slot + ((lane>>4)<<4)
// (contiguous K order, matching the B fragment loads).
__global__ __launch_bounds__(256)
void packw_kernel(const float* __restrict__ ew, const float* __restrict__ rbuf,
                  unsigned short* __restrict__ wpack) {
  const size_t total = (size_t)16 * 9 * 8 * 16 * 32 * 16;   // 9,437,184
  for (size_t idx = (size_t)blockIdx.x * 256 + threadIdx.x; idx < total;
       idx += (size_t)gridDim.x * 256) {
    const int slot = (int)(idx & 15);
    const int lane = (int)((idx >> 4) & 31);
    const int MT   = (int)((idx >> 9) & 15);
    const int cc   = (int)((idx >> 13) & 7);
    const int bt   = (int)(idx >> 16);
    const int t = bt % 9, b = bt / 9;
    const int o = MT * 16 + (lane & 15);
    const int kk = slot + ((lane >> 4) << 4);
    const int c = cc * 32 + kk;
    float acc = 0.f;
    #pragma unroll
    for (int k = 0; k < 3; ++k)
      acc += rbuf[((size_t)b * 3 + k) * 256 + c] *
             ew[(((size_t)k * 256 + o) * 256 + c) * 9 + t];
    wpack[idx] = f2bf(acc);
  }
}

// ------------------------------ Phase 3: implicit-GEMM conv via WMMA --------
// Block = 8 waves. Block tile: one sample b, one output row h, 128 c_out x
// 64 pixels. Wave (mp = wave>>1, np = wave&1): 2Mx2N register tile =
// 32 c_out x 32 pixels, 4 f32 accumulators; A and B fragments each reused
// twice -> 8 b128 loads per 4 WMMAs. K loop: 9 taps x 8 chunks of 32 c_in.
__global__ __launch_bounds__(256)
void gemm_kernel(const unsigned short* __restrict__ xp_,
                 const unsigned short* __restrict__ wpack_,
                 float* __restrict__ out) {
  const __bf16* __restrict__ xp = (const __bf16*)xp_;
  const __bf16* __restrict__ wp = (const __bf16*)wpack_;

  const int mt128 = blockIdx.x & 1;                // c_out block of 128
  const int h     = (blockIdx.x >> 1) & 63;        // output row
  const int b     = blockIdx.x >> 7;               // sample
  const int lane = threadIdx.x & 31;
  const int wave = threadIdx.x >> 5;
  const int mp = wave >> 1;                        // M-tile pair 0..3
  const int np = wave & 1;                         // N-tile pair 0..1
  const int l15 = lane & 15;
  const int hiK = lane >> 4;
  const int MTg0 = mt128 * 8 + mp * 2;             // first of 2 global M tiles

  // A: wpack offset(t,cc) = (((b*9+t)*8+cc)*16 + MTg)*512 + lane*16 halfs
  const __bf16* abase = wp + ((size_t)b * 9 * 8 * 16 + MTg0) * 512 + lane * 16;
  // B: xp[b][yp][wpix][c]; lane holds pixel col l15, K run = 16 consecutive c
  const int bcol = hiK * 16;                       // + cc*32 added in loop
  const int p0 = np * 32 + l15;                    // first N-tile pixel

  v8f acc00 = {}; v8f acc01 = {}; v8f acc10 = {}; v8f acc11 = {};

  for (int t = 0; t < 9; ++t) {
    const int dy = t / 3, dx = t % 3;
    const __bf16* xrow = xp + ((size_t)(b * 66 + (h + dy))) * 66 * 256;
    const __bf16* bpA = xrow + (size_t)(p0 + dx) * 256 + bcol;        // nt0
    const __bf16* bpB = bpA + (size_t)16 * 256;                       // nt1
    const __bf16* ap = abase + (size_t)(t * 8) * 16 * 512;
    #pragma unroll 2
    for (int cc = 0; cc < 8; ++cc) {
      BF16Frag a0, a1, b0, b1;
      a0.h[0] = *(const v8bf*)(ap);
      a0.h[1] = *(const v8bf*)(ap + 8);
      a1.h[0] = *(const v8bf*)(ap + 512);        // adjacent M tile
      a1.h[1] = *(const v8bf*)(ap + 520);
      b0.h[0] = *(const v8bf*)(bpA);
      b0.h[1] = *(const v8bf*)(bpA + 8);
      b1.h[0] = *(const v8bf*)(bpB);
      b1.h[1] = *(const v8bf*)(bpB + 8);
      acc00 = __builtin_amdgcn_wmma_f32_16x16x32_bf16(
          false, a0.v, false, b0.v, (short)0, acc00, false, false);
      acc01 = __builtin_amdgcn_wmma_f32_16x16x32_bf16(
          false, a0.v, false, b1.v, (short)0, acc01, false, false);
      acc10 = __builtin_amdgcn_wmma_f32_16x16x32_bf16(
          false, a1.v, false, b0.v, (short)0, acc10, false, false);
      acc11 = __builtin_amdgcn_wmma_f32_16x16x32_bf16(
          false, a1.v, false, b1.v, (short)0, acc11, false, false);
      ap  += 16 * 512;   // next c-chunk (16 M-tiles apart)
      bpA += 32;         // next 32 input channels
      bpB += 32;
    }
  }

  // C/D layout: lane<16 -> N=lane, VGPR r -> M=r; lane>=16 -> M=r+8.
  const int obase = (mt128 << 7) + (mp << 5) + (hiK << 3);
  const int w0 = (np << 5) + l15;
  float* ob = out + ((size_t)b * 256 + obase) * 4096 + (size_t)h * 64;
  #pragma unroll
  for (int r = 0; r < 8; ++r) {
    ob[(size_t)r * 4096 + w0]             = acc00[r];
    ob[(size_t)r * 4096 + w0 + 16]        = acc01[r];
    ob[(size_t)(r + 16) * 4096 + w0]      = acc10[r];
    ob[(size_t)(r + 16) * 4096 + w0 + 16] = acc11[r];
  }
}

// ---------------------------------------------------------------------------
extern "C" void kernel_launch(void* const* d_in, const int* in_sizes, int n_in,
                              void* d_out, int out_size, void* d_ws, size_t ws_size,
                              hipStream_t stream) {
  const float* x       = (const float*)d_in[0];    // (16,256,64,64)
  const float* ew      = (const float*)d_in[1];    // (3,256,256,3,3)
  const float* rconv_w = (const float*)d_in[2];    // (16,256)
  const float* rconv_b = (const float*)d_in[3];    // (16,)
  const float* fc_w    = (const float*)d_in[4];    // (768,256)
  const float* fc_b    = (const float*)d_in[5];    // (768,)
  float* out = (float*)d_out;                      // (16,256,64,64)

  char* ws = (char*)d_ws;
  float*          pooled = (float*)(ws);                       //   262,144 B
  float*          rbuf   = (float*)(ws + 262144);              //    49,152 B
  unsigned short* xp     = (unsigned short*)(ws + 327680);     // 35,684,352 B
  unsigned short* wpk    = (unsigned short*)(ws + 36012032);   // 18,874,368 B
  const size_t xp_bytes = (size_t)16 * 66 * 66 * 256 * 2;

  pool_kernel <<<16 * 256, 256, 0, stream>>>(x, pooled);
  route_kernel<<<16,       256, 0, stream>>>(pooled, rconv_w, rconv_b, fc_w, fc_b, rbuf);
  zero_kernel <<<4352,     256, 0, stream>>>((uint4*)xp, xp_bytes / 16);
  packx_kernel<<<16 * 64,  256, 0, stream>>>(x, xp);
  packw_kernel<<<4608,     256, 0, stream>>>(ew, rbuf, wpk);
  gemm_kernel <<<16 * 64 * 2, 256, 0, stream>>>(xp, wpk, out);
}
